// VectorField_21414706937934
// MI455X (gfx1250) — compile-verified, matrix operand order checked by reading, not run on hardware
//
#include <hip/hip_runtime.h>
#include <cstdint>
#include <cstddef>

typedef __attribute__((ext_vector_type(16))) _Float16 v16h;
typedef __attribute__((ext_vector_type(8)))  _Float16 v8h;
typedef __attribute__((ext_vector_type(4)))  _Float16 v4h;
typedef __attribute__((ext_vector_type(8)))  float    v8f;
typedef __attribute__((ext_vector_type(4)))  int      v4i;

#define B_ROWS 65536
#define D_IN   768
#define H_DIM  512
#define L_LVL  16
#define T_SIZE 524288          // 2^19
#define IN_DIM 832             // D + L*4

#if defined(__has_builtin)
#  if __has_builtin(__builtin_amdgcn_global_load_async_to_lds_b128) && \
      __has_builtin(__builtin_amdgcn_s_wait_asynccnt)
#    define USE_ASYNC_LDS 1
#  endif
#endif
#ifndef USE_ASYNC_LDS
#  define USE_ASYNC_LDS 0
#endif

#if USE_ASYNC_LDS
// Issue one 16-byte-per-lane async copy global->LDS (GLOBAL_LOAD_ASYNC_TO_LDS
// _B128, tracked by ASYNCcnt). Signature (from clang): (int4 AS1*, int4 AS3*,
// imm offset, imm cpol). Generic LDS pointers truncate to the AS3 offset in
// their low 32 bits (ISA aperture rule), so inttoptr casts are safe.
__device__ __forceinline__ void async_cp16(const void* g, const void* l) {
  __builtin_amdgcn_global_load_async_to_lds_b128(
      (__attribute__((address_space(1))) v4i*)(uintptr_t)g,
      (__attribute__((address_space(3))) v4i*)(uint32_t)(uintptr_t)l,
      0, 0);
}
#endif

// floor(16 * 32^(l/15)) = floor(16 * 2^(l/3)), matches the reference's scales
__device__ __constant__ float c_scales[16] = {
    16.f, 20.f, 25.f, 32.f, 40.f, 50.f, 64.f, 80.f,
    101.f, 128.f, 161.f, 203.f, 256.f, 322.f, 406.f, 512.f};

// ---------------------------------------------------------------------------
// Weight packing: f32 [K,N] row-major -> f16 fragment-packed layout.
// For k-tile kt, column n: 32 halves at ((kt*N + n) * 32 + p(k&31)), where
// p() reorders K so positions 0..15 hold K{0..7,16..23} (lanes 0-15) and
// positions 16..31 hold K{8..15,24..31} — the 16-bit B-operand lane layout of
// v_wmma_f32_16x16x32_f16.
// ---------------------------------------------------------------------------
__global__ void pack_w_kernel(const float* __restrict__ W,
                              _Float16* __restrict__ Wp, int K, int N) {
  int gid = blockIdx.x * blockDim.x + threadIdx.x;   // K*N threads
  int k = gid / N, n = gid - k * N;
  int kr = k & 31;
  int p = (kr & 7) | ((kr & 8) << 1) | ((kr & 16) >> 1);
  size_t dst = ((size_t)((k >> 5) * N + n) << 5) + (size_t)p;
  Wp[dst] = (_Float16)W[gid];
}

// x (f32) -> first 768 cols of the fused f16 activation matrix (stride 832)
__global__ void cvt_x_kernel(const float* __restrict__ x,
                             _Float16* __restrict__ A0) {
  int gid = blockIdx.x * blockDim.x + threadIdx.x;   // B * 192 threads
  int row = gid / (D_IN / 4);
  int c4  = gid - row * (D_IN / 4);
  float4 v = ((const float4*)x)[gid];
  v4h h;
  h[0] = (_Float16)v.x; h[1] = (_Float16)v.y;
  h[2] = (_Float16)v.z; h[3] = (_Float16)v.w;
  *(v4h*)(A0 + (size_t)row * IN_DIM + c4 * 4) = h;
}

// Multi-res hash encode: one thread per (row, level); 8 float4 corner gathers
// (hot table lives in the 192MB L2), trilinear blend, 4 f16 features out.
__global__ void hash_encode_kernel(const float* __restrict__ t,
                                   const float* __restrict__ cond,
                                   const float* __restrict__ tables,
                                   _Float16* __restrict__ A0) {
  int gid = blockIdx.x * blockDim.x + threadIdx.x;   // B * 16 threads
  int b = gid >> 4, l = gid & 15;
  float s  = c_scales[l];
  float xf0 = cond[2 * b] * s;
  float xf1 = cond[2 * b + 1] * s;
  float xf2 = t[b] * s;
  float f0 = floorf(xf0), f1 = floorf(xf1), f2 = floorf(xf2);
  float w0 = xf0 - f0, w1 = xf1 - f1, w2 = xf2 - f2;
  uint32_t i0 = (uint32_t)(int)f0, i1 = (uint32_t)(int)f1, i2 = (uint32_t)(int)f2;
  const float4* tab = (const float4*)tables + (size_t)l * T_SIZE;
  float a0 = 0.f, a1 = 0.f, a2 = 0.f, a3 = 0.f;
#pragma unroll
  for (int c = 0; c < 8; ++c) {
    uint32_t o0 = (c >> 2) & 1, o1 = (c >> 1) & 1, o2 = c & 1;
    uint32_t h = (i0 + o0) ^ ((i1 + o1) * 2654435761u) ^ ((i2 + o2) * 805459861u);
    float4 f = tab[h & (uint32_t)(T_SIZE - 1)];
    float wc = (o0 ? w0 : 1.f - w0) * (o1 ? w1 : 1.f - w1) * (o2 ? w2 : 1.f - w2);
    a0 += wc * f.x; a1 += wc * f.y; a2 += wc * f.z; a3 += wc * f.w;
  }
  v4h h4;
  h4[0] = (_Float16)a0; h4[1] = (_Float16)a1;
  h4[2] = (_Float16)a2; h4[3] = (_Float16)a3;
  *(v4h*)(A0 + (size_t)b * IN_DIM + D_IN + l * 4) = h4;
}

__device__ __forceinline__ v16h frag_cat(v8h lo, v8h hi) {
  return __builtin_shufflevector(lo, hi, 0, 1, 2, 3, 4, 5, 6, 7,
                                 8, 9, 10, 11, 12, 13, 14, 15);
}

// ---------------------------------------------------------------------------
// WMMA GEMM: out = [silu](A[M,K]f16 @ W[K,N]f16 + bias).
// 128x128 block tile, K stepped 32; 8 waves each own 64x32 = 4x2 grid of
// v_wmma_f32_16x16x32_f16. Double-buffered LDS; the next k-tile is fetched
// with GLOBAL_LOAD_ASYNC_TO_LDS_B128 (ASYNCcnt) directly into the alternate
// buffer while the current buffer feeds the WMMAs; one barrier per k-step.
// ---------------------------------------------------------------------------
template <bool SILU>
__global__ __launch_bounds__(256)
void gemm_wmma_kernel(const _Float16* __restrict__ A,
                      const _Float16* __restrict__ Wp,
                      const float* __restrict__ bias,
                      void* __restrict__ outv, int N, int K) {
  constexpr int LDSS = 40;               // halves per row: 32 + 8 pad (80B)
  constexpr int BUFH = 128 * LDSS;       // halves per buffer
  __shared__ _Float16 As[2][BUFH];
  __shared__ _Float16 Bs[2][BUFH];

  const int tid  = threadIdx.x;
  const int lane = tid & 31;
  const int wave = tid >> 5;
  const int wm   = (wave & 1) * 64;      // wave M offset inside block tile
  const int wn   = (wave >> 1) * 32;     // wave N offset inside block tile
  const int m0   = blockIdx.x * 128;
  const int n0   = blockIdx.y * 128;

  const int ldr = tid >> 1;              // 0..127 (tile row / packed column)
  const int ldc = (tid & 1) * 16;        // 0 or 16 halves

  const int mrow  = lane & 15;
  const int khalf = (lane >> 4) * 8;     // K phase: lanes 0-15 vs 16-31

  const _Float16* agBase = A  + (size_t)(m0 + ldr) * K + ldc;
  const _Float16* bgBase = Wp + ((size_t)(n0 + ldr) << 5) + ldc;
  const size_t    bgStep = (size_t)N << 5;       // halves per k-tile of Wp

  _Float16* asDst = &As[0][ldr * LDSS + ldc];
  _Float16* bsDst = &Bs[0][ldr * LDSS + ldc];

#if USE_ASYNC_LDS
#define ISSUE_TILE(kt_, buf_)                                                  \
  do {                                                                         \
    const _Float16* ag_ = agBase + ((kt_) << 5);                               \
    const _Float16* bg_ = bgBase + (size_t)(kt_)*bgStep;                       \
    _Float16* ad_ = asDst + (buf_)*BUFH;                                       \
    _Float16* bd_ = bsDst + (buf_)*BUFH;                                       \
    async_cp16(ag_, ad_);                                                      \
    async_cp16(ag_ + 8, ad_ + 8);                                              \
    async_cp16(bg_, bd_);                                                      \
    async_cp16(bg_ + 8, bd_ + 8);                                              \
  } while (0)
#define WAIT_TILE() __builtin_amdgcn_s_wait_asynccnt(0)
#else
#define ISSUE_TILE(kt_, buf_)                                                  \
  do {                                                                         \
    const _Float16* ag_ = agBase + ((kt_) << 5);                               \
    const _Float16* bg_ = bgBase + (size_t)(kt_)*bgStep;                       \
    uint4 a0_ = *(const uint4*)ag_;                                            \
    uint4 a1_ = *(const uint4*)(ag_ + 8);                                      \
    uint4 b0_ = *(const uint4*)bg_;                                            \
    uint4 b1_ = *(const uint4*)(bg_ + 8);                                      \
    _Float16* ad_ = asDst + (buf_)*BUFH;                                       \
    _Float16* bd_ = bsDst + (buf_)*BUFH;                                       \
    *(uint4*)ad_ = a0_;                                                        \
    *(uint4*)(ad_ + 8) = a1_;                                                  \
    *(uint4*)bd_ = b0_;                                                        \
    *(uint4*)(bd_ + 8) = b1_;                                                  \
  } while (0)
#define WAIT_TILE() ((void)0)
#endif

  v8f acc[4][2] = {};
  const int KT = K >> 5;

  ISSUE_TILE(0, 0);
  WAIT_TILE();
  __syncthreads();

  for (int kt = 0; kt < KT; ++kt) {
    const int cur = kt & 1;
    if (kt + 1 < KT) ISSUE_TILE(kt + 1, 1 - cur);  // fill alternate buffer

    const _Float16* asb = &As[cur][0];
    const _Float16* bsb = &Bs[cur][0];
    v16h af[4], bf[2];
#pragma unroll
    for (int mt = 0; mt < 4; ++mt) {
      const _Float16* p = asb + (wm + mt * 16 + mrow) * LDSS;
      af[mt] = frag_cat(*(const v8h*)(p + khalf), *(const v8h*)(p + khalf + 16));
    }
#pragma unroll
    for (int nt = 0; nt < 2; ++nt) {
      const _Float16* p = bsb + (wn + nt * 16 + mrow) * LDSS + (khalf << 1);
      bf[nt] = frag_cat(*(const v8h*)p, *(const v8h*)(p + 8));
    }
#pragma unroll
    for (int mt = 0; mt < 4; ++mt)
#pragma unroll
      for (int nt = 0; nt < 2; ++nt)
        acc[mt][nt] = __builtin_amdgcn_wmma_f32_16x16x32_f16(
            false, af[mt], false, bf[nt], (short)0, acc[mt][nt], false, false);

    if (kt + 1 < KT) {
      WAIT_TILE();        // next tile resident in LDS (this wave's copies)
      __syncthreads();    // visible to all waves; cur buffer fully consumed
    }
  }

  // Epilogue: bias (+ fast SiLU via v_rcp_f32 + f16 cast) per the 16x16 f32
  // C/D lane layout (VGPR r: lanes 0-15 -> M=r, lanes 16-31 -> M=8+r).
  const int mhi = (lane >> 4) * 8;
#pragma unroll
  for (int nt = 0; nt < 2; ++nt) {
    int n = n0 + wn + nt * 16 + mrow;
    float bv = bias[n];
#pragma unroll
    for (int mt = 0; mt < 4; ++mt) {
#pragma unroll
      for (int r = 0; r < 8; ++r) {
        int m = m0 + wm + mt * 16 + mhi + r;
        float v = acc[mt][nt][r] + bv;
        if (SILU) {
          v = v * __builtin_amdgcn_rcpf(1.0f + __expf(-v));
          ((_Float16*)outv)[(size_t)m * N + n] = (_Float16)v;
        } else {
          ((float*)outv)[(size_t)m * N + n] = v;
        }
      }
    }
  }
#undef ISSUE_TILE
#undef WAIT_TILE
}

// ---------------------------------------------------------------------------
extern "C" void kernel_launch(void* const* d_in, const int* in_sizes, int n_in,
                              void* d_out, int out_size, void* d_ws, size_t ws_size,
                              hipStream_t stream) {
  const float* t      = (const float*)d_in[0];
  const float* x      = (const float*)d_in[1];
  const float* cond   = (const float*)d_in[2];
  const float* tables = (const float*)d_in[3];
  const float* W0 = (const float*)d_in[4];
  const float* b0 = (const float*)d_in[5];
  const float* W1 = (const float*)d_in[6];
  const float* b1 = (const float*)d_in[7];
  const float* W2 = (const float*)d_in[8];
  const float* b2 = (const float*)d_in[9];
  const float* W3 = (const float*)d_in[10];
  const float* b3 = (const float*)d_in[11];

  char*  ws  = (char*)d_ws;
  size_t off = 0;
  auto carve = [&](size_t bytes) -> char* {
    char* p = ws + off;
    off += bytes;
    off = (off + 255) & ~(size_t)255;
    return p;
  };
  _Float16* Wp0 = (_Float16*)carve((size_t)IN_DIM * H_DIM * 2);
  _Float16* Wp1 = (_Float16*)carve((size_t)H_DIM * H_DIM * 2);
  _Float16* Wp2 = (_Float16*)carve((size_t)H_DIM * H_DIM * 2);
  _Float16* Wp3 = (_Float16*)carve((size_t)H_DIM * D_IN * 2);
  _Float16* A0  = (_Float16*)carve((size_t)B_ROWS * IN_DIM * 2);  // 832-wide
  _Float16* P   = (_Float16*)carve((size_t)B_ROWS * H_DIM * 2);   // 512-wide
  _Float16* Q   = A0;  // A0 region reused once layer-0 has consumed it

  // Weight repack (tiny)
  pack_w_kernel<<<(IN_DIM * H_DIM) / 256, 256, 0, stream>>>(W0, Wp0, IN_DIM, H_DIM);
  pack_w_kernel<<<(H_DIM * H_DIM) / 256, 256, 0, stream>>>(W1, Wp1, H_DIM, H_DIM);
  pack_w_kernel<<<(H_DIM * H_DIM) / 256, 256, 0, stream>>>(W2, Wp2, H_DIM, H_DIM);
  pack_w_kernel<<<(H_DIM * D_IN) / 256, 256, 0, stream>>>(W3, Wp3, H_DIM, D_IN);

  // Fused activation build: [x | hash_encode(cond,t)] as f16, stride 832
  cvt_x_kernel<<<(B_ROWS * (D_IN / 4)) / 256, 256, 0, stream>>>(x, A0);
  hash_encode_kernel<<<(B_ROWS * L_LVL) / 256, 256, 0, stream>>>(t, cond, tables, A0);

  // MLP: silu(A0@W0+b0) -> silu(@W1+b1) -> silu(@W2+b2) -> @W3+b3 (f32 out)
  dim3 blk(256);
  gemm_wmma_kernel<true ><<<dim3(B_ROWS / 128, H_DIM / 128), blk, 0, stream>>>(A0, Wp0, b0, P, H_DIM, IN_DIM);
  gemm_wmma_kernel<true ><<<dim3(B_ROWS / 128, H_DIM / 128), blk, 0, stream>>>(P,  Wp1, b1, Q, H_DIM, H_DIM);
  gemm_wmma_kernel<true ><<<dim3(B_ROWS / 128, H_DIM / 128), blk, 0, stream>>>(Q,  Wp2, b2, P, H_DIM, H_DIM);
  gemm_wmma_kernel<false><<<dim3(B_ROWS / 128, D_IN / 128), blk, 0, stream>>>(P,  Wp3, b3, d_out, D_IN, H_DIM);
}
